// ald_fev8_onlycentre_conti_18305150616067
// MI455X (gfx1250) — matrix operand order repaired and checked by reading, not compiled, not run on hardware
//
#include <hip/hip_runtime.h>
#include <hip/hip_bf16.h>

typedef __attribute__((ext_vector_type(16))) _Float16 v16h;
typedef __attribute__((ext_vector_type(8)))  float    v8f;

#define NEGC (-100000.0f)

__device__ __constant__ int c_FULL_PERMS[3][16] = {
  {0,1,2,3,4,5,6,7, 8,9,10,11,12,13,14,15},
  {4,5,6,7,8,9,10,11, 0,1,2,3,12,13,14,15},
  {8,9,10,11,12,13,14,15, 0,1,2,3,4,5,6,7}
};

// ---------------------------------------------------------------------------
// K1: fea_map (B,1024,28,28) -> 16-cell pooled sums -> area argmax -> perms
// ---------------------------------------------------------------------------
__global__ void k_area(const float* __restrict__ fea, int* __restrict__ perms,
                       int* __restrict__ inv, float* __restrict__ out_perms) {
  const int b = blockIdx.x, t = threadIdx.x;
  __shared__ float cells[16];
  if (t < 16) cells[t] = 0.f;
  __syncthreads();
  if (t < 784) {
    const int y = t / 28, xx = t % 28;
    const float* p = fea + (size_t)b * 1024u * 784u + t;
    float acc = 0.f;
    for (int ch = 0; ch < 1024; ++ch) acc += p[(size_t)ch * 784u];
    atomicAdd(&cells[(y / 7) * 4 + (xx / 7)], acc);   // ds_add_f32
  }
  __syncthreads();
  if (t == 0) {
    float s0 = 0.f, s1 = 0.f, s2 = 0.f;
    for (int k = 0; k < 8; ++k) { s0 += cells[k]; s1 += cells[4 + k]; s2 += cells[8 + k]; }
    int area = 0; float best = s0;
    if (s1 > best) { best = s1; area = 1; }
    if (s2 > best) { best = s2; area = 2; }
    for (int k = 0; k < 16; ++k) {
      const int pk = c_FULL_PERMS[area][k];
      perms[b * 16 + k]  = pk;
      inv[b * 16 + pk]   = k;
      out_perms[b * 16 + k] = (float)pk;
    }
  }
}

// ---------------------------------------------------------------------------
// K2: peel_sum. One 1024-thread block per batch (32x32 matrix, 1 elem/thread).
// ---------------------------------------------------------------------------
__global__ void k_peel(const float* __restrict__ pc, const int* __restrict__ bseq,
                       const int* __restrict__ bsz, float* __restrict__ rout) {
  const int b = blockIdx.x, tid = threadIdx.x;
  const int i = tid >> 5, j = tid & 31;
  __shared__ float m[1024], red[1024], red2[1024], mx[1024], sc[1024];
  __shared__ float s_max, s_sec, s_bmax;

  const int start = bseq[0] * bsz[0];
  const float v = pc[(size_t)(start + b) * 1024u + tid];

  // softmax over axis=1 (rows i, per column j)
  red[tid] = v; __syncthreads();
  for (int s = 16; s; s >>= 1) { if (i < s) red[tid] = fmaxf(red[tid], red[tid + (s << 5)]); __syncthreads(); }
  const float cmax = red[j]; __syncthreads();
  const float e = expf(v - cmax);
  red[tid] = e; __syncthreads();
  for (int s = 16; s; s >>= 1) { if (i < s) red[tid] += red[tid + (s << 5)]; __syncthreads(); }
  const float csum = red[j]; __syncthreads();
  m[tid] = e / csum;
  float acc = 0.f;
  __syncthreads();

  for (int iter = 0; iter < 32; ++iter) {
    const float val = m[tid];
    // global top-2 via (max,sec) merge tree
    mx[tid] = val; sc[tid] = -3.4e38f; __syncthreads();
    for (int s = 512; s; s >>= 1) {
      if (tid < s) {
        const float am = mx[tid], bm = mx[tid + s];
        const float as_ = sc[tid], bs_ = sc[tid + s];
        mx[tid] = fmaxf(am, bm);
        sc[tid] = fmaxf(fminf(am, bm), fmaxf(as_, bs_));
      }
      __syncthreads();
    }
    if (tid == 0) { s_max = mx[0]; s_sec = sc[0]; }
    __syncthreads();
    const float mid = 0.5f * (s_max + s_sec), hd = 0.5f * (s_max - s_sec);
    float bv = fmaxf(val - mid, 0.f) / hd;
    // b / max(b)
    mx[tid] = bv; __syncthreads();
    for (int s = 512; s; s >>= 1) { if (tid < s) mx[tid] = fmaxf(mx[tid], mx[tid + s]); __syncthreads(); }
    if (tid == 0) s_bmax = mx[0];
    __syncthreads();
    bv = bv / s_bmax;
    // row sums (over j) and column sums (over i)
    red[tid] = bv; red2[tid] = bv; __syncthreads();
    for (int s = 16; s; s >>= 1) {
      if (j < s) red[tid]  += red[tid + s];
      if (i < s) red2[tid] += red2[tid + (s << 5)];
      __syncthreads();
    }
    const float rs = red[i << 5], cs = red2[j];
    __syncthreads();
    m[tid] = val + NEGC * (rs + cs);
    acc += bv;
    __syncthreads();
  }
  rout[(size_t)b * 1024u + tid] = acc;
}

// ---------------------------------------------------------------------------
// K2b: batch-reverse r, write result_centre (x3) to d_out + f16 copy for WMMA.
// ---------------------------------------------------------------------------
__global__ void k_post(const float* __restrict__ rout, float* __restrict__ out_rc,
                       _Float16* __restrict__ rch, int B) {
  const int bo = blockIdx.x, tid = threadIdx.x;
  const float v = rout[(size_t)(B - 1 - bo) * 1024u + tid];
  out_rc[((size_t)bo * 3 + 0) * 1024u + tid] = v;
  out_rc[((size_t)bo * 3 + 1) * 1024u + tid] = v;
  out_rc[((size_t)bo * 3 + 2) * 1024u + tid] = v;
  rch[(size_t)bo * 1024u + tid] = (_Float16)v;
}

// ---------------------------------------------------------------------------
// K3: fused gather -> WMMA (16x16x32 f16, f32 accum) -> coalesced scatter.
// One wave per 16-row M-tile, looping over the 3 channels (B fragments and
// perm table reused). Gather/scatter lanes follow consecutive rows p, so each
// half-wave touches 64B-contiguous global segments. LDS staging is f32,
// bank-conflict-free ([kk][row] for A, padded [M][j] stride 33 for D).
// ---------------------------------------------------------------------------
__global__ __launch_bounds__(32) void k_mat(const float* __restrict__ x,
                                            const _Float16* __restrict__ rch,
                                            const int* __restrict__ perms,
                                            float* __restrict__ xout) {
  const int tile = blockIdx.x;   // 0..195 (3136/16)
  const int b    = blockIdx.y;
  const int lane = threadIdx.x;  // 0..31, one wave32
  const int p0 = tile * 16;
  const int r  = lane & 15;      // row-within-tile for gather/store loops (=M)
  const int hi = lane >> 4;

  __shared__ float lds_a[32 * 16];   // [kk][row] f32
  __shared__ float lds_d[16 * 33];   // [M][j] padded
  __shared__ int sp[8];
  if (lane < 8) sp[lane] = perms[b * 16 + lane];

  // row-address pieces shared by gather and scatter: p = p0 + r
  const int p  = p0 + r;
  const int a_ = p / 56, bb = p - a_ * 56;

  // B fragments once per block (rc identical across channels).
  // Lane N = lane&15; low half-wave K 0..15, high half-wave K 16..31.
  const _Float16* rb = rch + (size_t)b * 1024u;
  const int N0 = lane & 15;
  v16h B0, B1;
  #pragma unroll
  for (int vg = 0; vg < 8; ++vg) {
    const int k = hi * 16 + 2 * vg;
    B0[2 * vg]     = rb[k * 32 + N0];
    B0[2 * vg + 1] = rb[(k + 1) * 32 + N0];
    B1[2 * vg]     = rb[k * 32 + 16 + N0];
    B1[2 * vg + 1] = rb[(k + 1) * 32 + 16 + N0];
  }

  for (int ch = 0; ch < 3; ++ch) {
    const float* xb = x    + ((size_t)b * 3 + ch) * (448u * 448u);
    float*       ob = xout + ((size_t)b * 3 + ch) * (448u * 448u);

    __syncthreads();   // sp ready (ch=0); lds_a/lds_d reusable (ch>0)

    // Gather: iteration it covers kk = 2*it + hi; lanes walk consecutive p.
    #pragma unroll
    for (int it = 0; it < 16; ++it) {
      const int kk = 2 * it + hi;
      const int r4 = kk >> 3, c8 = kk & 7;
      const int yr = r4 * 56 + a_, yc = c8 * 56 + bb;
      const int khi = (yr >= 112) ? 1 : 0;
      const int n = sp[khi * 4 + (yc / 112)];
      const int row = (n >> 2) * 112 + (yr - khi * 112);
      const int col = (n & 3) * 112 + (yc % 112);
      lds_a[kk * 16 + r] = xb[row * 448 + col];
    }
    __syncthreads();

    // A fragment: lane M = lane&15;
    // low half-wave K 0..15, high half-wave K 16..31 (matches B).
    v16h A;
    #pragma unroll
    for (int vg = 0; vg < 8; ++vg) {
      const int kb = hi * 16 + 2 * vg;
      A[2 * vg]     = (_Float16)lds_a[kb * 16 + r];
      A[2 * vg + 1] = (_Float16)lds_a[(kb + 1) * 16 + r];
    }

    v8f C0 = {}; v8f C1 = {};
    C0 = __builtin_amdgcn_wmma_f32_16x16x32_f16(false, A, false, B0, (short)0, C0, false, false);
    C1 = __builtin_amdgcn_wmma_f32_16x16x32_f16(false, A, false, B1, (short)0, C1, false, false);

    // Stage D in LDS: lane N = lane&15, VGPR vg -> M = vg + 8*hi.
    #pragma unroll
    for (int vg = 0; vg < 8; ++vg) {
      const int Mr = vg + 8 * hi;
      lds_d[Mr * 33 + N0]      = C0[vg];
      lds_d[Mr * 33 + 16 + N0] = C1[vg];
    }
    __syncthreads();

    // Coalesced scatter: iteration it covers j = 2*it + hi; lanes walk p.
    #pragma unroll
    for (int it = 0; it < 16; ++it) {
      const int jj = 2 * it + hi;
      const int r4 = jj >> 3, c8 = jj & 7;
      const int yr = r4 * 56 + a_, yc = c8 * 56 + bb;
      const int khi = (yr >= 112) ? 1 : 0;
      const int n = sp[khi * 4 + (yc / 112)];
      const int row = (n >> 2) * 112 + (yr - khi * 112);
      const int col = (n & 3) * 112 + (yc % 112);
      ob[row * 448 + col] = lds_d[r * 33 + jj];
    }
  }
}

// ---------------------------------------------------------------------------
// K4: boundary patches pass through unchanged (inv[n] >= 8).
// ---------------------------------------------------------------------------
__global__ void k_bound(const float* __restrict__ x, const int* __restrict__ inv,
                        float* __restrict__ xout, int total) {
  const int idx = blockIdx.x * blockDim.x + threadIdx.x;
  if (idx >= total) return;
  const int pix = idx % (448 * 448);
  const int b   = idx / (3 * 448 * 448);
  const int row = pix / 448, col = pix % 448;
  const int n = (row / 112) * 4 + (col / 112);
  if (inv[b * 16 + n] >= 8) xout[idx] = x[idx];
}

// ---------------------------------------------------------------------------
extern "C" void kernel_launch(void* const* d_in, const int* in_sizes, int n_in,
                              void* d_out, int out_size, void* d_ws, size_t ws_size,
                              hipStream_t stream) {
  const float* x   = (const float*)d_in[0];            // (B,3,448,448)
  const float* fea = (const float*)d_in[1];            // (B,1024,28,28)
  const float* pc  = (const float*)d_in[2];            // (5000,32,32)
  const int* bseq  = (const int*)d_in[3];
  const int* bsz   = (const int*)d_in[4];

  const int B = in_sizes[0] / (3 * 448 * 448);         // 64
  const size_t xtot = (size_t)B * 3 * 448 * 448;       // 38,535,168
  float* out = (float*)d_out;
  float* out_rc    = out + xtot;                        // (B,3,32,32)
  float* out_perms = out + xtot + (size_t)B * 3 * 1024; // (B,16)

  // workspace layout (byte offsets)
  char* ws = (char*)d_ws;
  int*      ws_perms = (int*)(ws);                                // B*16 ints
  int*      ws_inv   = (int*)(ws + 4096);                         // B*16 ints
  float*    ws_r     = (float*)(ws + 8192);                       // B*32*32 f32
  _Float16* ws_rch   = (_Float16*)(ws + 8192 + (size_t)B * 4096); // B*32*32 f16

  k_area<<<B, 1024, 0, stream>>>(fea, ws_perms, ws_inv, out_perms);
  k_peel<<<B, 1024, 0, stream>>>(pc, bseq, bsz, ws_r);
  k_post<<<B, 1024, 0, stream>>>(ws_r, out_rc, ws_rch, B);
  k_mat<<<dim3(196, B), 32, 0, stream>>>(x, ws_rch, ws_perms, out);
  const int total = (int)xtot;
  k_bound<<<(total + 255) / 256, 256, 0, stream>>>(x, ws_inv, out, total);
}
